// MCTDFMN_29901562314897
// MI455X (gfx1250) — matrix operand order repaired and checked by reading, not compiled, not run on hardware
//
#include <hip/hip_runtime.h>

// ---------------- problem constants ----------------
#define KC   64        // classes
#define SSUP 5         // shots per class
#define CCH  640       // channels
#define HH   5         // spatial
#define DD   16000     // C*H*H feature dim
#define QQN  2000      // queries
#define QP   2016      // queries padded to 63*32 (WMMA needs full tiles, EXEC all-1s)
#define NIT  10        // refinement iterations
#define BNEPS 2e-5f

typedef __attribute__((ext_vector_type(16))) __bf16 bvec16;
typedef __attribute__((ext_vector_type(8)))  float  fvec8;

// load 16 bf16 as two contiguous 8-elem (16B) runs -> A/B fragment register layout
__device__ __forceinline__ bvec16 ld_frag2(const __bf16* p0, const __bf16* p1) {
    union { uint4 u[2]; bvec16 v; } c;
    c.u[0] = *reinterpret_cast<const uint4*>(p0);
    c.u[1] = *reinterpret_cast<const uint4*>(p1);
    return c.v;
}
__device__ __forceinline__ bvec16 ld_frag16(const __bf16* p) {   // 16 contiguous bf16
    return ld_frag2(p, p + 8);
}

// ---------------- sup_sum + initial protos ----------------
__global__ __launch_bounds__(256)
void k_sum_support(const float* __restrict__ sup, float* __restrict__ sup_sum,
                   float* __restrict__ protos) {
    int idx = blockIdx.x * 256 + threadIdx.x;
    if (idx >= KC * DD) return;
    int k = idx / DD, d = idx - k * DD;
    const float* p = sup + (size_t)k * SSUP * DD + d;
    float s = 0.f;
#pragma unroll
    for (int i = 0; i < SSUP; ++i) s += p[(size_t)i * DD];
    sup_sum[idx] = s;
    protos[idx]  = s * (1.0f / SSUP);
}

// ---------------- ScaleModule + normalize + bf16 pack ----------------
// one block per sample; computes conv3x3(640->1,VALID 3x3) -> BN -> ReLU -> fc -> softplus = s
// writes ofeat = x/(norm*s) in bf16, oinv2 = 1/s^2 (== ||ofeat||^2), orawscale = norm*s
__global__ __launch_bounds__(256)
void k_pack_scale(const float* __restrict__ xin, int n_valid,
                  const float* __restrict__ cw, const float* __restrict__ cb,
                  const float* __restrict__ bg, const float* __restrict__ bb,
                  const float* __restrict__ bm, const float* __restrict__ bv,
                  const float* __restrict__ fw, const float* __restrict__ fb,
                  __bf16* __restrict__ ofeat, float* __restrict__ oinv2,
                  float* __restrict__ orawscale) {
    int s = blockIdx.x, t = threadIdx.x;
    __bf16* orow = ofeat + (size_t)s * DD;
    if (s >= n_valid) {   // zero padding rows so WMMA tiles over QP are exact
        for (int d = t; d < DD; d += 256) orow[d] = (__bf16)0.0f;
        if (t == 0) { oinv2[s] = 0.f; if (orawscale) orawscale[s] = 0.f; }
        return;
    }
    const float* x = xin + (size_t)s * DD;
    __shared__ float red[256];
    __shared__ float shc[9];
    __shared__ float shv[2];

    float ss = 0.f;
    for (int d = t; d < DD; d += 256) { float v = x[d]; ss += v * v; }

    float a9[9];
#pragma unroll
    for (int p = 0; p < 9; ++p) a9[p] = 0.f;
    for (int c = t; c < CCH; c += 256) {
        const float* xb = x + c * 25;
        const float* wb = cw + c * 9;
        float wv[9], xv[25];
#pragma unroll
        for (int i = 0; i < 9; ++i)  wv[i] = wb[i];
#pragma unroll
        for (int i = 0; i < 25; ++i) xv[i] = xb[i];
#pragma unroll
        for (int oi = 0; oi < 3; ++oi)
#pragma unroll
            for (int oj = 0; oj < 3; ++oj) {
                float acc = 0.f;
#pragma unroll
                for (int di = 0; di < 3; ++di)
#pragma unroll
                    for (int dj = 0; dj < 3; ++dj)
                        acc += xv[(oi + di) * 5 + oj + dj] * wv[di * 3 + dj];
                a9[oi * 3 + oj] += acc;
            }
    }
    // reduce sum-of-squares
    red[t] = ss; __syncthreads();
    for (int h = 128; h > 0; h >>= 1) { if (t < h) red[t] += red[t + h]; __syncthreads(); }
    if (t == 0) shv[1] = red[0];
    __syncthreads();
    // reduce 9 conv outputs
    for (int p = 0; p < 9; ++p) {
        red[t] = a9[p]; __syncthreads();
        for (int h = 128; h > 0; h >>= 1) { if (t < h) red[t] += red[t + h]; __syncthreads(); }
        if (t == 0) shc[p] = red[0];
        __syncthreads();
    }
    if (t == 0) {
        float inv = bg[0] * rsqrtf(bv[0] + BNEPS);
        float z = fb[0];
#pragma unroll
        for (int p = 0; p < 9; ++p) {
            float y = shc[p] + cb[0];
            y = (y - bm[0]) * inv + bb[0];
            y = fmaxf(y, 0.f);
            z += y * fw[p];
        }
        float sp = (z > 20.f) ? z : ((z < -20.f) ? __expf(z) : log1pf(__expf(z)));
        float nrm = fmaxf(sqrtf(fmaxf(shv[1], 0.f)), 1e-12f);
        shv[0] = 1.0f / (nrm * sp);
        oinv2[s] = 1.0f / (sp * sp);
        if (orawscale) orawscale[s] = nrm * sp;
    }
    __syncthreads();
    float inv = shv[0];
    for (int d = t; d < DD; d += 256) orow[d] = (__bf16)(x[d] * inv);
}

// ---------------- distances: out[q,k] = -(qq[q]+pp[k]-2*qf.pf) via WMMA ----------------
// block = 8 waves covering 32q x 64k; grid = QP/32 = 63
__global__ __launch_bounds__(256)
void k_dist_wmma(const __bf16* __restrict__ qf, const __bf16* __restrict__ pf,
                 const float* __restrict__ qq, const float* __restrict__ pp,
                 float* __restrict__ out, int qmax) {
    int w = threadIdx.x >> 5, lane = threadIdx.x & 31;
    int qbase = blockIdx.x * 32 + (w >> 2) * 16;
    int kbase = (w & 3) * 16;
    int m  = lane & 15;     // A row (q) / B col (k)
    int kh = lane >> 4;
    const __bf16* arow = qf + (size_t)(qbase + m) * DD;
    const __bf16* brow = pf + (size_t)(kbase + m) * DD;
    fvec8 acc = {0.f, 0.f, 0.f, 0.f, 0.f, 0.f, 0.f, 0.f};
    for (int dc = 0; dc < DD; dc += 32) {
        bvec16 a = ld_frag16(arow + dc + kh * 16);
        bvec16 b = ld_frag16(brow + dc + kh * 16);
        acc = __builtin_amdgcn_wmma_f32_16x16x32_bf16(false, a, false, b,
                                                      (short)0, acc, false, false);
    }
    int k = kbase + m;
    float pk = pp[k];
#pragma unroll
    for (int r = 0; r < 8; ++r) {
        int q = qbase + r + 8 * kh;
        if (q < qmax) out[(size_t)q * KC + k] = -(qq[q] + pk - 2.0f * acc[r]);
    }
}

__global__ void k_zero64(float* c) { if (threadIdx.x < KC) c[threadIdx.x] = 0.f; }

// ---------------- softmax rows, fold rawscale, build pwT [64 x QP], colsum ----------------
__global__ __launch_bounds__(256)
void k_softmax_pw(const float* __restrict__ logits, const float* __restrict__ rawscale,
                  __bf16* __restrict__ pwT, float* __restrict__ colsum) {
    __shared__ float cs[KC];
    int t = threadIdx.x, w = t >> 5, lane = t & 31;
    if (t < KC) cs[t] = 0.f;
    __syncthreads();
    int q = blockIdx.x * 8 + w;
    if (q < QQN) {
        float x0 = logits[(size_t)q * KC + lane];
        float x1 = logits[(size_t)q * KC + lane + 32];
        float mx = fmaxf(x0, x1);
        for (int off = 16; off; off >>= 1) mx = fmaxf(mx, __shfl_xor(mx, off, 32));
        float e0 = __expf(x0 - mx), e1 = __expf(x1 - mx);
        float s = e0 + e1;
        for (int off = 16; off; off >>= 1) s += __shfl_xor(s, off, 32);
        float inv = 1.0f / s;
        float p0 = e0 * inv, p1 = e1 * inv;
        float rs = rawscale[q];
        pwT[(size_t)lane * QP + q]        = (__bf16)(p0 * rs);
        pwT[(size_t)(lane + 32) * QP + q] = (__bf16)(p1 * rs);
        atomicAdd(&cs[lane], p0);
        atomicAdd(&cs[lane + 32], p1);
    } else if (q < QP) {   // padded rows contribute nothing to the update GEMM
        pwT[(size_t)lane * QP + q]        = (__bf16)0.0f;
        pwT[(size_t)(lane + 32) * QP + q] = (__bf16)0.0f;
    }
    __syncthreads();
    if (t < KC) atomicAdd(&colsum[t], cs[t]);
}

// ---------------- proto update: protos[k,d]=(sup_sum + pwT^T x qf)/(S+colsum[k]) ----------------
// block = 8 waves covering 64k x 32d; grid = DD/32 = 500; qf tile transposed through LDS
__global__ __launch_bounds__(256)
void k_update_wmma(const __bf16* __restrict__ pwT, const __bf16* __restrict__ qf,
                   const float* __restrict__ sup_sum, const float* __restrict__ colsum,
                   float* __restrict__ protos) {
    __shared__ __align__(16) __bf16 ldsB[32 * 32];   // [d_local][q_local]
    int t = threadIdx.x, w = t >> 5, lane = t & 31;
    int dbase = blockIdx.x * 32;
    int mbase = (w >> 1) * 16;       // class-tile
    int ntile = (w & 1) * 16;        // d-subtile
    int m = lane & 15, kh = lane >> 4;
    int ql = t >> 3, dg = t & 7;     // staging roles
    const __bf16* arow = pwT + (size_t)(mbase + m) * QP;
    fvec8 acc = {0.f, 0.f, 0.f, 0.f, 0.f, 0.f, 0.f, 0.f};
    for (int qc = 0; qc < QP; qc += 32) {
        union { uint2 u; __bf16 h[4]; } ld;
        ld.u = *reinterpret_cast<const uint2*>(qf + (size_t)(qc + ql) * DD + dbase + dg * 4);
        __syncthreads();             // prior-iteration LDS reads done
#pragma unroll
        for (int i = 0; i < 4; ++i) ldsB[(dg * 4 + i) * 32 + ql] = ld.h[i];
        __syncthreads();
        bvec16 a = ld_frag16(arow + qc + kh * 16);
        bvec16 b = ld_frag16(&ldsB[(ntile + m) * 32 + kh * 16]);
        acc = __builtin_amdgcn_wmma_f32_16x16x32_bf16(false, a, false, b,
                                                      (short)0, acc, false, false);
    }
    int d = dbase + ntile + m;
#pragma unroll
    for (int r = 0; r < 8; ++r) {
        int kc = mbase + r + 8 * kh;
        float denom = (float)SSUP + colsum[kc];
        protos[(size_t)kc * DD + d] = (sup_sum[(size_t)kc * DD + d] + acc[r]) / denom;
    }
}

// ---------------- host orchestration ----------------
extern "C" void kernel_launch(void* const* d_in, const int* in_sizes, int n_in,
                              void* d_out, int out_size, void* d_ws, size_t ws_size,
                              hipStream_t stream) {
    const float* sup = (const float*)d_in[0];
    const float* qry = (const float*)d_in[1];
    const float* cw  = (const float*)d_in[2];
    const float* cb  = (const float*)d_in[3];
    const float* bg  = (const float*)d_in[4];
    const float* bb  = (const float*)d_in[5];
    const float* bm  = (const float*)d_in[6];
    const float* bv  = (const float*)d_in[7];
    const float* fw  = (const float*)d_in[8];
    const float* fb  = (const float*)d_in[9];
    float* out = (float*)d_out;

    char* ws = (char*)d_ws;
    size_t off = 0;
    auto alloc = [&](size_t bytes) -> void* {
        void* p = ws + off;
        off = (off + bytes + 255) & ~(size_t)255;
        return p;
    };
    __bf16* qf      = (__bf16*)alloc((size_t)QP * DD * 2);
    __bf16* pf      = (__bf16*)alloc((size_t)KC * DD * 2);
    __bf16* pwT     = (__bf16*)alloc((size_t)KC * QP * 2);
    float*  logits  = (float*) alloc((size_t)QP * KC * 4);
    float*  protos  = (float*) alloc((size_t)KC * DD * 4);
    float*  sup_sum = (float*) alloc((size_t)KC * DD * 4);
    float*  qq      = (float*) alloc((size_t)QP * 4);
    float*  rsc     = (float*) alloc((size_t)QP * 4);
    float*  pp      = (float*) alloc((size_t)KC * 4);
    float*  colsum  = (float*) alloc((size_t)KC * 4);

    k_sum_support<<<(KC * DD + 255) / 256, 256, 0, stream>>>(sup, sup_sum, protos);
    k_pack_scale<<<QP, 256, 0, stream>>>(qry, QQN, cw, cb, bg, bb, bm, bv, fw, fb,
                                         qf, qq, rsc);
    for (int it = 0; it < NIT; ++it) {
        k_pack_scale<<<KC, 256, 0, stream>>>(protos, KC, cw, cb, bg, bb, bm, bv, fw, fb,
                                             pf, pp, nullptr);
        k_dist_wmma<<<QP / 32, 256, 0, stream>>>(qf, pf, qq, pp, logits, QP);
        k_zero64<<<1, 64, 0, stream>>>(colsum);
        k_softmax_pw<<<QP / 8, 256, 0, stream>>>(logits, rsc, pwT, colsum);
        k_update_wmma<<<DD / 32, 256, 0, stream>>>(pwT, qf, sup_sum, colsum, protos);
    }
    k_pack_scale<<<KC, 256, 0, stream>>>(protos, KC, cw, cb, bg, bb, bm, bv, fw, fb,
                                         pf, pp, nullptr);
    k_dist_wmma<<<QP / 32, 256, 0, stream>>>(qf, pf, qq, pp, out, QQN);
}
